// ClotGRUUnrolled_43997644980835
// MI455X (gfx1250) — matrix operand order repaired and checked by reading, not compiled, not run on hardware
//
#include <hip/hip_runtime.h>
#include <hip/hip_bf16.h>

typedef __attribute__((ext_vector_type(16))) __bf16 v16bf;
typedef __attribute__((ext_vector_type(8)))  float  v8f;
typedef int v4i __attribute__((vector_size(16)));

#define B_   512
#define T_   128
#define I_   128
#define H_   768
#define G3H  2304   // 3*H
#define F1_  256
#define C_   10
#define NBLK 144    // 18 col tiles x 8 row tiles (persistent grid)

#if __has_builtin(__builtin_amdgcn_global_load_async_to_lds_b128)
#define USE_ASYNC_LDS 1
#else
#define USE_ASYNC_LDS 0
#endif

#if USE_ASYNC_LDS
__device__ __forceinline__ void async_cp_b128(const __bf16* g, __bf16* l)
{
    // (global src int4*, LDS dst int4*, imm offset, imm cpol)
    __builtin_amdgcn_global_load_async_to_lds_b128(
        (__attribute__((address_space(1))) v4i*)(g),
        (__attribute__((address_space(3))) v4i*)(l), 0, 0);
}
__device__ __forceinline__ void async_wait0()
{
#if __has_builtin(__builtin_amdgcn_s_wait_asynccnt)
    __builtin_amdgcn_s_wait_asynccnt(0);
#else
    asm volatile("s_wait_asynccnt 0" ::: "memory");
#endif
}
#endif

// ---------------------------------------------------------------------------
// 64x128-tile bf16 WMMA GEMM body:  C[M0..+64][N0..+128] = A * W^T (+bias,relu)
// A: rows over M, K cols; bf16 (stride lda) or f32 (converted while staging).
// W: [N][K] bf16 row-major (B-matrix = W^T).
// 256 threads = 8 waves in 2x4; each wave: 32x32 tile = 2x2 accumulators,
// K-step 32 via v_wmma_f32_16x16x32_bf16.
// Pipeline: tile k+1 transfer issued before computing tile k; LDS double
// buffer -> one barrier per K-iteration. Transfers use async global->LDS DMA
// (ASYNCcnt) when the toolchain provides the builtin, else VGPR staging.
// ---------------------------------------------------------------------------
template <bool AF32, bool HAS_BIAS, bool RELU>
__device__ __forceinline__ void gemm_tile(
    const void* Aptr, int lda,
    const __bf16* __restrict__ W, int ldw,
    int K,
    float* __restrict__ C, int ldc,
    const float* __restrict__ bias,
    int M0, int N0,
    __bf16 (*As)[64][40], __bf16 (*Bs)[128][40])
{
    const int tid  = threadIdx.x;
    const int lane = tid & 31;
    const int hf   = lane >> 4;
    const int l16  = lane & 15;
    const int wave = tid >> 5;
    const int wm   = (wave >> 2) * 32;   // 0 or 32
    const int wn   = (wave & 3) * 32;    // 0..96

    const __bf16* Ab = (const __bf16*)Aptr;
    const float*  Af = (const float*)Aptr;

    v8f acc[2][2];
#pragma unroll
    for (int i = 0; i < 2; ++i)
#pragma unroll
        for (int j = 0; j < 2; ++j)
#pragma unroll
            for (int e = 0; e < 8; ++e) acc[i][j][e] = 0.0f;

    const int  srow = tid >> 1;          // B: 0..127 ; A uses tid<128 -> 0..63
    const int  skk  = (tid & 1) * 16;
    const bool aAct = tid < 128;         // waves 0-3 stage A (uniform per wave)

    uint4  ra0, ra1, rb0, rb1;           // staged bf16 tiles (fallback path)
    float4 fa0, fa1, fa2, fa3;           // staged f32 A tile (AF32 path)

    // Issue transfer of tile k0 targeting LDS buffer bufT.
    auto loadT = [&](int k0, int bufT) {
        if (aAct) {
            if (AF32) {
                const float* s = Af + (size_t)(M0 + srow) * lda + k0 + skk;
                fa0 = ((const float4*)s)[0];
                fa1 = ((const float4*)s)[1];
                fa2 = ((const float4*)s)[2];
                fa3 = ((const float4*)s)[3];
            } else {
#if USE_ASYNC_LDS
                const __bf16* s = Ab + (size_t)(M0 + srow) * lda + k0 + skk;
                async_cp_b128(s,     &As[bufT][srow][skk]);
                async_cp_b128(s + 8, &As[bufT][srow][skk + 8]);
#else
                const uint4* s = (const uint4*)(Ab + (size_t)(M0 + srow) * lda + k0 + skk);
                ra0 = s[0];
                ra1 = s[1];
#endif
            }
        }
        {
            const __bf16* s = W + (size_t)(N0 + srow) * ldw + k0 + skk;
#if USE_ASYNC_LDS
            async_cp_b128(s,     &Bs[bufT][srow][skk]);
            async_cp_b128(s + 8, &Bs[bufT][srow][skk + 8]);
#else
            const uint4* sv = (const uint4*)s;
            rb0 = sv[0];
            rb1 = sv[1];
#endif
        }
        (void)bufT;
    };
    // Register -> LDS stores (everything already in LDS on the async path
    // except the converted f32 A tile).
    auto storeT = [&](int buf) {
        if (aAct && AF32) {
            __bf16* d = &As[buf][srow][skk];
            d[0]=(__bf16)fa0.x; d[1]=(__bf16)fa0.y; d[2]=(__bf16)fa0.z; d[3]=(__bf16)fa0.w;
            d[4]=(__bf16)fa1.x; d[5]=(__bf16)fa1.y; d[6]=(__bf16)fa1.z; d[7]=(__bf16)fa1.w;
            d[8]=(__bf16)fa2.x; d[9]=(__bf16)fa2.y; d[10]=(__bf16)fa2.z; d[11]=(__bf16)fa2.w;
            d[12]=(__bf16)fa3.x; d[13]=(__bf16)fa3.y; d[14]=(__bf16)fa3.z; d[15]=(__bf16)fa3.w;
        }
#if !USE_ASYNC_LDS
        if (aAct && !AF32) {
            uint4* d = (uint4*)&As[buf][srow][skk];
            d[0] = ra0;
            d[1] = ra1;
        }
        {
            uint4* d = (uint4*)&Bs[buf][srow][skk];
            d[0] = rb0;
            d[1] = rb1;
        }
#endif
        (void)buf;
    };

    loadT(0, 0);
    int buf = 0;
    for (int k0 = 0; k0 < K; k0 += 32) {
        storeT(buf);
#if USE_ASYNC_LDS
        async_wait0();                     // tile-k DMA into `buf` complete
#endif
        __syncthreads();
        if (k0 + 32 < K) loadT(k0 + 32, buf ^ 1);   // in flight during compute

        v16bf afrag[2];
#pragma unroll
        for (int i = 0; i < 2; ++i) {
            const __bf16* ap = &As[buf][wm + i * 16 + l16][0];
#pragma unroll
            for (int e = 0; e < 8; ++e) afrag[i][e]     = ap[8 * hf + e];
#pragma unroll
            for (int e = 0; e < 8; ++e) afrag[i][8 + e] = ap[16 + 8 * hf + e];
        }
        v16bf bfrag[2];
#pragma unroll
        for (int j = 0; j < 2; ++j) {
            const __bf16* bp = &Bs[buf][wn + j * 16 + l16][0];
#pragma unroll
            for (int e = 0; e < 16; ++e) bfrag[j][e] = bp[(e < 8) ? (8 * hf + e) : (16 + 8 * hf + (e - 8))];
        }
#pragma unroll
        for (int i = 0; i < 2; ++i)
#pragma unroll
            for (int j = 0; j < 2; ++j)
                acc[i][j] = __builtin_amdgcn_wmma_f32_16x16x32_bf16(
                    false, afrag[i], false, bfrag[j], (short)0, acc[i][j],
                    false, false);
        buf ^= 1;
    }

    // ---- epilogue: C/D layout M = vgpr + 8*half, N = lane&15 ----
#pragma unroll
    for (int j = 0; j < 2; ++j) {
        const int col = N0 + wn + j * 16 + l16;
        float bv = 0.0f;
        if (HAS_BIAS) bv = bias[col];
#pragma unroll
        for (int i = 0; i < 2; ++i) {
            float* p = C + (size_t)(M0 + wm + i * 16 + 8 * hf) * ldc + col;
#pragma unroll
            for (int r = 0; r < 8; ++r) {
                float v = acc[i][j][r];
                if (HAS_BIAS) v += bv;
                if (RELU) v = fmaxf(v, 0.0f);
                *p = v;
                p += ldc;
            }
        }
    }
}

// ---------------------------------------------------------------------------
// Device-wide barrier: monotonically increasing counter (zeroed per launch).
// ---------------------------------------------------------------------------
__device__ __forceinline__ void grid_barrier(unsigned* bar, unsigned nb, unsigned& epoch)
{
    epoch += 1;  // uniform across all threads/blocks
    __syncthreads();
    if (threadIdx.x == 0) {
        __hip_atomic_fetch_add(bar, 1u, __ATOMIC_ACQ_REL, __HIP_MEMORY_SCOPE_AGENT);
        const unsigned target = epoch * nb;
        while (__hip_atomic_load(bar, __ATOMIC_ACQUIRE, __HIP_MEMORY_SCOPE_AGENT) < target) {
            __builtin_amdgcn_s_sleep(2);
        }
    }
    __syncthreads();
}

// ---------------------------------------------------------------------------
// Persistent recurrent kernel: all 128 GRU steps in one launch.
// Phase A: Gh = h@w_hh^T (K=768) and Gi = x_t@w_ih^T (K=128) per 64x128 tile.
// Phase B (after grid barrier): elementwise gates, update h (f32 + bf16).
// ---------------------------------------------------------------------------
__global__ __launch_bounds__(256) void gru_persistent_kernel(
    const float* __restrict__ x,
    const __bf16* __restrict__ whh, const __bf16* __restrict__ wih,
    const float* __restrict__ bih, const float* __restrict__ bhh,
    float* __restrict__ h, __bf16* __restrict__ hbf,
    float* __restrict__ Gh, float* __restrict__ Gi,
    unsigned* bar)
{
    __shared__ __bf16 As[2][64][40];
    __shared__ __bf16 Bs[2][128][40];

    const int bid = blockIdx.x;
    const int N0  = (bid % 18) * 128;   // gate-column tile
    const int M0  = (bid / 18) * 64;    // batch tile
    unsigned epoch = 0;

    for (int t = 0; t < T_; ++t) {
        gemm_tile<false, false, false>(hbf, H_, whh, H_, H_,
                                       Gh, G3H, nullptr, M0, N0, As, Bs);
        __syncthreads();
        gemm_tile<true, false, false>(x + (size_t)t * I_, T_ * I_, wih, I_, I_,
                                      Gi, G3H, nullptr, M0, N0, As, Bs);
        grid_barrier(bar, NBLK, epoch);

        for (int idx = bid * 256 + threadIdx.x; idx < B_ * H_; idx += NBLK * 256) {
            int b = idx / H_;
            int j = idx - b * H_;
            const float* gi = Gi + (size_t)b * G3H;
            const float* gh = Gh + (size_t)b * G3H;
            float pr = gi[j] + bih[j] + gh[j] + bhh[j];
            float pz = gi[H_ + j] + bih[H_ + j] + gh[H_ + j] + bhh[H_ + j];
            float r = 1.0f / (1.0f + __expf(-pr));
            float z = 1.0f / (1.0f + __expf(-pz));
            float n = tanhf(gi[2 * H_ + j] + bih[2 * H_ + j] +
                            r * (gh[2 * H_ + j] + bhh[2 * H_ + j]));
            float hn = (1.0f - z) * n + z * h[idx];
            h[idx] = hn;
            hbf[idx] = (__bf16)hn;
        }
        grid_barrier(bar, NBLK, epoch);
    }
}

__global__ __launch_bounds__(256) void fc1_gemm_kernel(
    const __bf16* __restrict__ hbf, const __bf16* __restrict__ wfc1,
    const float* __restrict__ bias, float* __restrict__ C)
{
    __shared__ __bf16 As[2][64][40];
    __shared__ __bf16 Bs[2][128][40];
    gemm_tile<false, true, true>(hbf, H_, wfc1, H_, H_, C, F1_, bias,
                                 blockIdx.y * 64, blockIdx.x * 128, As, Bs);
}

__global__ __launch_bounds__(256) void cvt_f32_bf16_kernel(
    const float* __restrict__ src, __bf16* __restrict__ dst, int n)
{
    int i = blockIdx.x * blockDim.x + threadIdx.x;
    if (i < n) dst[i] = (__bf16)src[i];
}

__global__ __launch_bounds__(256) void init_kernel(
    float* __restrict__ h, __bf16* __restrict__ hbf, unsigned* __restrict__ bar)
{
    int i = blockIdx.x * blockDim.x + threadIdx.x;   // < B*H
    h[i] = 0.0f;
    hbf[i] = (__bf16)0.0f;
    if (i == 0) *bar = 0u;
}

__global__ __launch_bounds__(256) void fc2_kernel(
    const float* __restrict__ a, const float* __restrict__ w2,
    const float* __restrict__ b2, float* __restrict__ out)
{
    int idx = blockIdx.x * blockDim.x + threadIdx.x;   // < B*C
    int b = idx / C_;
    int c = idx - b * C_;
    float s = b2[c];
    const float* ar = a + (size_t)b * F1_;
    const float* wr = w2 + (size_t)c * F1_;
#pragma unroll 8
    for (int k = 0; k < F1_; ++k) s += ar[k] * wr[k];
    out[idx] = s;
}

// ---------------------------------------------------------------------------
// Workspace layout (bytes)
// ---------------------------------------------------------------------------
#define OFF_WHH   ((size_t)0)
#define OFF_WIH   (OFF_WHH  + (size_t)G3H*H_*2)
#define OFF_WFC1  (OFF_WIH  + (size_t)G3H*I_*2)
#define OFF_H     (OFF_WFC1 + (size_t)F1_*H_*2)
#define OFF_HBF   (OFF_H    + (size_t)B_*H_*4)
#define OFF_GH    (OFF_HBF  + (size_t)B_*H_*2)
#define OFF_GI    (OFF_GH   + (size_t)B_*G3H*4)
#define OFF_FC1   (OFF_GI   + (size_t)B_*G3H*4)
#define OFF_BAR   (OFF_FC1  + (size_t)B_*F1_*4)

extern "C" void kernel_launch(void* const* d_in, const int* in_sizes, int n_in,
                              void* d_out, int out_size, void* d_ws, size_t ws_size,
                              hipStream_t stream)
{
    (void)in_sizes; (void)n_in; (void)out_size; (void)ws_size;

    const float* x     = (const float*)d_in[0];
    const float* w_ih  = (const float*)d_in[1];
    const float* w_hh  = (const float*)d_in[2];
    const float* b_ih  = (const float*)d_in[3];
    const float* b_hh  = (const float*)d_in[4];
    const float* w_fc1 = (const float*)d_in[5];
    const float* b_fc1 = (const float*)d_in[6];
    const float* w_fc2 = (const float*)d_in[7];
    const float* b_fc2 = (const float*)d_in[8];
    float* out = (float*)d_out;

    char* ws = (char*)d_ws;
    __bf16*   whh_bf  = (__bf16*)(ws + OFF_WHH);
    __bf16*   wih_bf  = (__bf16*)(ws + OFF_WIH);
    __bf16*   wfc1_bf = (__bf16*)(ws + OFF_WFC1);
    float*    h       = (float*)(ws + OFF_H);
    __bf16*   hbf     = (__bf16*)(ws + OFF_HBF);
    float*    Gh      = (float*)(ws + OFF_GH);
    float*    Gi      = (float*)(ws + OFF_GI);
    float*    fc1o    = (float*)(ws + OFF_FC1);
    unsigned* bar     = (unsigned*)(ws + OFF_BAR);

    // --- prep: bf16 weight copies, h0 = 0, barrier counter = 0 ---
    cvt_f32_bf16_kernel<<<(G3H * H_ + 255) / 256, 256, 0, stream>>>(w_hh, whh_bf, G3H * H_);
    cvt_f32_bf16_kernel<<<(G3H * I_ + 255) / 256, 256, 0, stream>>>(w_ih, wih_bf, G3H * I_);
    cvt_f32_bf16_kernel<<<(F1_ * H_ + 255) / 256, 256, 0, stream>>>(w_fc1, wfc1_bf, F1_ * H_);
    init_kernel<<<(B_ * H_) / 256, 256, 0, stream>>>(h, hbf, bar);

    // --- all 128 recurrent steps in one persistent launch ---
    gru_persistent_kernel<<<NBLK, 256, 0, stream>>>(
        x, whh_bf, wih_bf, b_ih, b_hh, h, hbf, Gh, Gi, bar);

    // --- head ---
    fc1_gemm_kernel<<<dim3(F1_ / 128, B_ / 64), 256, 0, stream>>>(hbf, wfc1_bf, b_fc1, fc1o);
    fc2_kernel<<<(B_ * C_) / 256, 256, 0, stream>>>(fc1o, w_fc2, b_fc2, out);
}
